// GAT_807453851657
// MI455X (gfx1250) — matrix operand order repaired and checked by reading, not compiled, not run on hardware
//
#include <hip/hip_runtime.h>
#include <hip/hip_bf16.h>

typedef __attribute__((ext_vector_type(16))) __bf16 v16bf;
typedef __attribute__((ext_vector_type(8)))  __bf16 v8bf;
typedef __attribute__((ext_vector_type(8)))  float  v8f;
typedef unsigned int u32x4 __attribute__((ext_vector_type(4)));
typedef int          i32x4 __attribute__((ext_vector_type(4)));
typedef int          i32x8 __attribute__((ext_vector_type(8)));

// ---- ordered-uint encoding for float atomicMax ----
static __device__ __forceinline__ unsigned enc_f32(float f) {
  unsigned u = __float_as_uint(f);
  return (u & 0x80000000u) ? ~u : (u | 0x80000000u);
}
static __device__ __forceinline__ float dec_f32(unsigned e) {
  unsigned u = (e & 0x80000000u) ? (e & 0x7fffffffu) : ~e;
  return __uint_as_float(u);
}
#define ENC_NEG_INF 0x007FFFFFu  // enc_f32(-inf)

// =====================================================================
// Pack fp32 weight W[K,Nc] (row-major) into bf16 WMMA-B layout, grouped
// so each 64-column group's tiles are CONTIGUOUS (one 1-D TDM copy):
//   offset = (((ntg*(K/32) + kt)*4 + t)*32 + lane)*16 + j
//   K index = kt*32 + 16*(lane>>4) + j ; N index = ntg*64 + t*16 + (lane&15)
// =====================================================================
__global__ void pack_b_kernel(const float* __restrict__ W, __bf16* __restrict__ out,
                              int K, int Nc) {
  int p = blockIdx.x * blockDim.x + threadIdx.x;
  if (p >= K * Nc) return;
  int j   = p & 15;
  int l   = (p >> 4) & 31;
  int t   = (p >> 9) & 3;
  int kts = K >> 5;
  int rest = p >> 11;
  int kt  = rest % kts;
  int ntg = rest / kts;
  int n = ntg * 64 + t * 16 + (l & 15);
  int k = kt * 32 + ((l >> 4) << 4) + j;
  out[p] = (__bf16)W[(size_t)k * Nc + n];
}

// =====================================================================
// WMMA GEMM: C[M,Nc] = A[M,K](bf16) * Bpacked + bias, optional ReLU.
// Requires K == 256, M >= 1. One wave -> 16x64 strip; the block's 32KB B
// column-group is staged global->LDS via the Tensor Data Mover and the
// WMMAs are fed from LDS (ds_load_b128).
// =====================================================================
__global__ __launch_bounds__(256) void wmma_gemm_bf16(
    const __bf16* __restrict__ A, const __bf16* __restrict__ Bp,
    const float* __restrict__ bias, float* __restrict__ Cf,
    __bf16* __restrict__ Cb, int M, int K, int Nc, int relu) {
  __shared__ __bf16 Bs[16384];  // (K/32)=8 ksteps * 4 tiles * 512 elems = 32 KB

  const int lane = threadIdx.x & 31;
  const int wave = threadIdx.x >> 5;
  const int row_base = (blockIdx.x * 8 + wave) * 16;
  const int col_base = blockIdx.y * 64;
  const int hi = lane >> 4;
  const int mlane = row_base + (lane & 15);
  const int kts = K >> 5;
  const size_t chunk = (size_t)K * 64;  // elements per 64-col group (16384)

  // Zero-fill Bs with REAL ds stores: without any visible store to the LDS
  // array, LLVM proves "non-escaping global, never written" and folds all B
  // reads to undef (the TDM builtin only takes integer descriptors).
  {
    i32x4 z = {0, 0, 0, 0};
    for (int i = threadIdx.x; i < (int)(chunk >> 3); i += 256)
      *(i32x4*)(Bs + (size_t)i * 8) = z;
  }
  __syncthreads();  // zero-fill complete before the TDM write lands

  // ---- stage this block's B column-group into LDS (TDM, TENSORcnt) ----
#if __has_builtin(__builtin_amdgcn_tensor_load_to_lds)
  if (threadIdx.x == 0) {
    unsigned long long ga =
        (unsigned long long)(size_t)(Bp + (size_t)blockIdx.y * chunk);
    unsigned nelem = (unsigned)chunk;  // 16384 (fits tile_dim0's 16 bits)
    u32x4 g0;
    g0.x = 1u;                                 // count=1: valid descriptor
    g0.y = 0u;                                 // lds_addr = 0 (Bs is sole LDS object)
    g0.z = (unsigned)ga;                       // global_addr[31:0]
    g0.w = (unsigned)(ga >> 32) | (2u << 30);  // global_addr[56:32] | type=2
    i32x8 g1;
    g1[0] = 1 << 16;                            // data_size = 1 (2-byte elements)
    g1[1] = (int)((nelem & 0xFFFFu) << 16);     // tensor_dim0[15:0]  (bits 63:48)
    g1[2] = (int)(nelem >> 16) | (1 << 16);     // tensor_dim0[31:16] | tensor_dim1=1
    g1[3] = (int)(nelem << 16);                 // tensor_dim1[31:16]=0 | tile_dim0
    g1[4] = 1;                                  // tile_dim1=1, tile_dim2=0
    g1[5] = (int)nelem;                         // tensor_dim0_stride[31:0]
    g1[6] = 0;                                  // stride0[47:32] | stride1[15:0]
    g1[7] = 0;                                  // stride1[47:16]
    i32x4 z4 = {0, 0, 0, 0};
#if __clang_major__ >= 23
    i32x8 z8 = {0, 0, 0, 0, 0, 0, 0, 0};
    __builtin_amdgcn_tensor_load_to_lds(g0, g1, z4, z4, z8, 0);
#else
    __builtin_amdgcn_tensor_load_to_lds(g0, g1, z4, z4, 0);
#endif
  }
  __builtin_amdgcn_s_wait_tensorcnt(0);  // issuing wave drains TENSORcnt
#else
  // fallback: cooperative global->LDS copy (same thread->address mapping as
  // the zero-fill, so no barrier needed in between beyond the one above)
  for (int i = threadIdx.x; i < (int)(chunk >> 3); i += 256)
    *(v8bf*)(Bs + (size_t)i * 8) =
        *(const v8bf*)(Bp + (size_t)blockIdx.y * chunk + (size_t)i * 8);
#endif
  // Escape Bs's address into an opaque asm that may write all memory: the
  // TDM writes LDS through a descriptor the compiler can't see, so this is
  // what forces the B reads below to be real ds loads (no zero-forwarding).
  {
    void* p = (void*)Bs;
    asm volatile("" : "+v"(p) :: "memory");
  }
  __syncthreads();

  v8f acc[4];
#pragma unroll
  for (int t = 0; t < 4; ++t)
#pragma unroll
    for (int r = 0; r < 8; ++r) acc[t][r] = 0.0f;

  // Row m of D depends only on row m of A: clamp OOB rows (their results
  // are never stored), so A loads are unconditional -> no exec juggling.
  const int mclamp = (mlane < M) ? mlane : (M - 1);
  const __bf16* arow = A + (size_t)mclamp * K + 16 * hi;

  for (int kt = 0; kt < kts; ++kt) {
    v16bf a;
    {
      const __bf16* ap = arow + kt * 32;
      v8bf lo  = *(const v8bf*)(ap);
      v8bf hic = *(const v8bf*)(ap + 8);
#pragma unroll
      for (int j = 0; j < 8; ++j) { a[j] = lo[j]; a[j + 8] = hic[j]; }
    }
#pragma unroll
    for (int t = 0; t < 4; ++t) {
      const __bf16* bp = Bs + (((kt * 4 + t) * 32 + lane) << 4);
      v8bf b0 = *(const v8bf*)(bp);
      v8bf b1 = *(const v8bf*)(bp + 8);
      v16bf b;
#pragma unroll
      for (int j = 0; j < 8; ++j) { b[j] = b0[j]; b[j + 8] = b1[j]; }
      acc[t] = __builtin_amdgcn_wmma_f32_16x16x32_bf16(
          false, a, false, b, (short)0, acc[t], false, false);
    }
  }

  // D layout: lanes 0-15: M=r, N=lane; lanes 16-31: M=8+r, N=lane-16
#pragma unroll
  for (int t = 0; t < 4; ++t) {
    int n = col_base + t * 16 + (lane & 15);
    float bv = bias ? bias[n] : 0.0f;
#pragma unroll
    for (int r = 0; r < 8; ++r) {
      int m = row_base + r + 8 * hi;
      if (m < M) {
        float v = acc[t][r] + bv;
        if (relu) v = fmaxf(v, 0.0f);
        if (Cf) Cf[(size_t)m * Nc + n] = v;
        if (Cb) Cb[(size_t)m * Nc + n] = (__bf16)v;
      }
    }
  }
}

// =====================================================================
// Layer 0 projections: x[N,2] -> Q/K/V (bf16) and Skip (f32), 256 ch.
// =====================================================================
__global__ void l0_proj_kernel(const float* __restrict__ x,
                               const float* __restrict__ Wq, const float* __restrict__ bq,
                               const float* __restrict__ Wk, const float* __restrict__ bk,
                               const float* __restrict__ Wv, const float* __restrict__ bv,
                               const float* __restrict__ Ws, const float* __restrict__ bs,
                               __bf16* __restrict__ Qb, __bf16* __restrict__ Kb,
                               __bf16* __restrict__ Vb, float* __restrict__ SkipF, int N) {
  int t = blockIdx.x * blockDim.x + threadIdx.x;
  if (t >= N * 256) return;
  int i = t >> 8, c = t & 255;
  float x0 = x[i * 2], x1 = x[i * 2 + 1];
  Qb[t] = (__bf16)(x0 * Wq[c] + x1 * Wq[256 + c] + bq[c]);
  Kb[t] = (__bf16)(x0 * Wk[c] + x1 * Wk[256 + c] + bk[c]);
  Vb[t] = (__bf16)(x0 * Wv[c] + x1 * Wv[256 + c] + bv[c]);
  SkipF[t] = x0 * Ws[c] + x1 * Ws[256 + c] + bs[c];
}

// ---- per-layer scratch init ----
__global__ void init_layer_kernel(float* __restrict__ agg, unsigned* __restrict__ nmax,
                                  float* __restrict__ nsum, int n256, int n4) {
  int t = blockIdx.x * blockDim.x + threadIdx.x;
  if (t < n256) agg[t] = 0.0f;
  if (t < n4) { nmax[t] = ENC_NEG_INF; nsum[t] = 0.0f; }
}

// =====================================================================
// Pass A: one wave per edge. alpha[e,h] = dot(q[dst], k[src]+ea*We)/8.
// 8-lane (=one head) shfl_xor reduction; atomicMax into encoded nmax.
// =====================================================================
__global__ __launch_bounds__(256) void edge_alpha_kernel(
    const __bf16* __restrict__ Qb, const __bf16* __restrict__ Kb,
    const float* __restrict__ ea, const float* __restrict__ We,
    const int* __restrict__ src, const int* __restrict__ dst,
    float* __restrict__ alpha, unsigned* __restrict__ nmax, int E) {
  int w = (blockIdx.x * blockDim.x + threadIdx.x) >> 5;
  int lane = threadIdx.x & 31;
  if (w >= E) return;
  int s = src[w], d = dst[w];
  float eav = ea[w];
  v8bf q8 = *(const v8bf*)(Qb + (size_t)d * 256 + lane * 8);
  v8bf k8 = *(const v8bf*)(Kb + (size_t)s * 256 + lane * 8);
  v8f  we = *(const v8f*)(We + lane * 8);
  float dot = 0.0f;
#pragma unroll
  for (int j = 0; j < 8; ++j)
    dot += (float)q8[j] * ((float)k8[j] + eav * we[j]);
#pragma unroll
  for (int off = 1; off < 8; off <<= 1)
    dot += __shfl_xor(dot, off, 32);
  if ((lane & 7) == 0) {
    int h = lane >> 3;
    float av = dot * 0.125f;  // 1/sqrt(64)
    alpha[(size_t)w * 4 + h] = av;
    atomicMax(&nmax[(size_t)d * 4 + h], enc_f32(av));
  }
}

// Pass B: exp(alpha - max) and segment sums.
__global__ void edge_exp_kernel(float* __restrict__ alpha, const unsigned* __restrict__ nmax,
                                float* __restrict__ nsum, const int* __restrict__ dst, int E) {
  int t = blockIdx.x * blockDim.x + threadIdx.x;
  if (t >= E * 4) return;
  int e = t >> 2, h = t & 3;
  int d = dst[e];
  float a = __expf(alpha[t] - dec_f32(nmax[(size_t)d * 4 + h]));
  alpha[t] = a;
  atomicAdd(&nsum[(size_t)d * 4 + h], a);
}

// Pass C: one wave per edge; scatter-add a/(sum)*(v+e) into agg[dst].
__global__ __launch_bounds__(256) void edge_aggr_kernel(
    const __bf16* __restrict__ Vb, const float* __restrict__ ea,
    const float* __restrict__ We, const float* __restrict__ alpha,
    const float* __restrict__ nsum, const int* __restrict__ src,
    const int* __restrict__ dst, float* __restrict__ agg, int E) {
  int w = (blockIdx.x * blockDim.x + threadIdx.x) >> 5;
  int lane = threadIdx.x & 31;
  if (w >= E) return;
  int s = src[w], d = dst[w];
  int h = lane >> 3;
  float coef = alpha[(size_t)w * 4 + h] / (nsum[(size_t)d * 4 + h] + 1e-16f);
  float eav = ea[w];
  v8bf v8 = *(const v8bf*)(Vb + (size_t)s * 256 + lane * 8);
  v8f  we = *(const v8f*)(We + lane * 8);
  float* out = agg + (size_t)d * 256 + lane * 8;
#pragma unroll
  for (int j = 0; j < 8; ++j)
    atomicAdd(out + j, coef * ((float)v8[j] + eav * we[j]));
}

// Epilogue: H = relu(agg + skip) -> bf16 activations for next layer.
__global__ void epilogue_relu_kernel(const float* __restrict__ agg,
                                     const float* __restrict__ skip,
                                     __bf16* __restrict__ Hb, int n) {
  int t = blockIdx.x * blockDim.x + threadIdx.x;
  if (t >= n) return;
  Hb[t] = (__bf16)fmaxf(agg[t] + skip[t], 0.0f);
}

// Classifier tail: out[i] = dot(Hc[i,0:128], W2) + b2 (wave per node).
__global__ __launch_bounds__(256) void cls_final_kernel(
    const __bf16* __restrict__ Hc, const float* __restrict__ W2,
    const float* __restrict__ b2, float* __restrict__ out, int N) {
  int w = (blockIdx.x * blockDim.x + threadIdx.x) >> 5;
  int lane = threadIdx.x & 31;
  if (w >= N) return;
  const __bf16* row = Hc + (size_t)w * 128 + lane * 4;
  float s = 0.0f;
#pragma unroll
  for (int j = 0; j < 4; ++j)
    s += (float)row[j] * W2[lane * 4 + j];
#pragma unroll
  for (int off = 1; off < 32; off <<= 1)
    s += __shfl_xor(s, off, 32);
  if (lane == 0) out[w] = s + b2[0];
}

// =====================================================================
extern "C" void kernel_launch(void* const* d_in, const int* in_sizes, int n_in,
                              void* d_out, int out_size, void* d_ws, size_t ws_size,
                              hipStream_t stream) {
  const int N = in_sizes[0] / 2;   // x: [N,2]
  const int E = in_sizes[1] / 2;   // edge_index: [2,E]

  const float* x      = (const float*)d_in[0];
  const int*   ei     = (const int*)d_in[1];
  const int*   src    = ei;
  const int*   dst    = ei + E;
  const float* ea     = (const float*)d_in[2];
  const float* l0_Wq  = (const float*)d_in[3];
  const float* l0_bq  = (const float*)d_in[4];
  const float* l0_Wk  = (const float*)d_in[5];
  const float* l0_bk  = (const float*)d_in[6];
  const float* l0_Wv  = (const float*)d_in[7];
  const float* l0_bv  = (const float*)d_in[8];
  const float* l0_We  = (const float*)d_in[9];
  const float* l0_Ws  = (const float*)d_in[10];
  const float* l0_bs  = (const float*)d_in[11];
  const float* r_Wq   = (const float*)d_in[12];
  const float* r_bq   = (const float*)d_in[13];
  const float* r_Wk   = (const float*)d_in[14];
  const float* r_bk   = (const float*)d_in[15];
  const float* r_Wv   = (const float*)d_in[16];
  const float* r_bv   = (const float*)d_in[17];
  const float* r_We   = (const float*)d_in[18];
  const float* r_Ws   = (const float*)d_in[19];
  const float* r_bs   = (const float*)d_in[20];
  const float* cls_W1 = (const float*)d_in[21];
  const float* cls_b1 = (const float*)d_in[22];
  const float* cls_W2 = (const float*)d_in[23];
  const float* cls_b2 = (const float*)d_in[24];

  // ---- workspace carve-up (256B aligned) ----
  char* ws = (char*)d_ws;
  size_t off = 0;
  auto alloc = [&](size_t bytes) -> void* {
    void* p = ws + off;
    off += (bytes + 255) & ~(size_t)255;
    return p;
  };
  __bf16*   Hb    = (__bf16*)alloc((size_t)N * 256 * 2);
  __bf16*   Qb    = (__bf16*)alloc((size_t)N * 256 * 2);
  __bf16*   Kb    = (__bf16*)alloc((size_t)N * 256 * 2);
  __bf16*   Vb    = (__bf16*)alloc((size_t)N * 256 * 2);
  float*    SkipF = (float*)alloc((size_t)N * 256 * 4);
  float*    AggF  = (float*)alloc((size_t)N * 256 * 4);
  float*    alpha = (float*)alloc((size_t)E * 4 * 4);
  unsigned* nmax  = (unsigned*)alloc((size_t)N * 4 * 4);
  float*    nsum  = (float*)alloc((size_t)N * 4 * 4);
  __bf16*   Wpk   = (__bf16*)alloc((size_t)8 * 256 * 256 * 2);  // 2 layers x {Wq,Wk,Wv,Ws}
  __bf16*   Wc1   = (__bf16*)alloc((size_t)256 * 128 * 2);
  __bf16*   Hc    = (__bf16*)alloc((size_t)N * 128 * 2);

  // ---- pack weights into grouped WMMA-B layout ----
  {
    int blk = 256, nb = (256 * 256 + blk - 1) / blk;
    for (int L = 0; L < 2; ++L) {
      pack_b_kernel<<<nb, blk, 0, stream>>>(r_Wq + L * 65536, Wpk + (size_t)(L * 4 + 0) * 65536, 256, 256);
      pack_b_kernel<<<nb, blk, 0, stream>>>(r_Wk + L * 65536, Wpk + (size_t)(L * 4 + 1) * 65536, 256, 256);
      pack_b_kernel<<<nb, blk, 0, stream>>>(r_Wv + L * 65536, Wpk + (size_t)(L * 4 + 2) * 65536, 256, 256);
      pack_b_kernel<<<nb, blk, 0, stream>>>(r_Ws + L * 65536, Wpk + (size_t)(L * 4 + 3) * 65536, 256, 256);
    }
    pack_b_kernel<<<(256 * 128 + blk - 1) / blk, blk, 0, stream>>>(cls_W1, Wc1, 256, 128);
  }

  // ---- layer 0 projections (K=2, memory-bound elementwise) ----
  l0_proj_kernel<<<(N * 256 + 255) / 256, 256, 0, stream>>>(
      x, l0_Wq, l0_bq, l0_Wk, l0_bk, l0_Wv, l0_bv, l0_Ws, l0_bs,
      Qb, Kb, Vb, SkipF, N);

  // ---- 3 TransformerConv layers ----
  for (int L = 0; L < 3; ++L) {
    const float* We = (L == 0) ? l0_We : (r_We + (size_t)(L - 1) * 256);

    init_layer_kernel<<<(N * 256 + 255) / 256, 256, 0, stream>>>(AggF, nmax, nsum, N * 256, N * 4);
    edge_alpha_kernel<<<(E * 32 + 255) / 256, 256, 0, stream>>>(Qb, Kb, ea, We, src, dst, alpha, nmax, E);
    edge_exp_kernel<<<(E * 4 + 255) / 256, 256, 0, stream>>>(alpha, nmax, nsum, dst, E);
    edge_aggr_kernel<<<(E * 32 + 255) / 256, 256, 0, stream>>>(Vb, ea, We, alpha, nsum, src, dst, AggF, E);
    epilogue_relu_kernel<<<(N * 256 + 255) / 256, 256, 0, stream>>>(AggF, SkipF, Hb, N * 256);

    if (L < 2) {  // next layer's Q/K/V/Skip via WMMA GEMM (256x256, bf16)
      dim3 grid((N + 127) / 128, 256 / 64);
      wmma_gemm_bf16<<<grid, 256, 0, stream>>>(Hb, Wpk + (size_t)(L * 4 + 0) * 65536, r_bq + L * 256,
                                               nullptr, Qb, N, 256, 256, 0);
      wmma_gemm_bf16<<<grid, 256, 0, stream>>>(Hb, Wpk + (size_t)(L * 4 + 1) * 65536, r_bk + L * 256,
                                               nullptr, Kb, N, 256, 256, 0);
      wmma_gemm_bf16<<<grid, 256, 0, stream>>>(Hb, Wpk + (size_t)(L * 4 + 2) * 65536, r_bv + L * 256,
                                               nullptr, Vb, N, 256, 256, 0);
      wmma_gemm_bf16<<<grid, 256, 0, stream>>>(Hb, Wpk + (size_t)(L * 4 + 3) * 65536, r_bs + L * 256,
                                               SkipF, nullptr, N, 256, 256, 0);
    }
  }

  // ---- classifier ----
  {
    dim3 grid((N + 127) / 128, 128 / 64);
    wmma_gemm_bf16<<<grid, 256, 0, stream>>>(Hb, Wc1, cls_b1, nullptr, Hc, N, 256, 128, 1);
    cls_final_kernel<<<(N * 32 + 255) / 256, 256, 0, stream>>>(Hc, cls_W2, cls_b2, (float*)d_out, N);
  }
}
